// Global_block3D_86569360818475
// MI455X (gfx1250) — matrix operand order repaired and checked by reading, not compiled, not run on hardware
//
#include <hip/hip_runtime.h>
#include <hip/hip_bf16.h>
#include <math.h>

typedef __bf16 bf16;
typedef __attribute__((ext_vector_type(8)))  __bf16 v8bf;
typedef __attribute__((ext_vector_type(16))) __bf16 v16bf;
typedef __attribute__((ext_vector_type(8)))  float  v8f;

// ---- static config (matches reference) ----
constexpr int cB = 4, cD = 8, cH = 56, cW = 56, cC = 192;
constexpr int cL = cD * cH * cW;                 // 25088 tokens per batch elem
constexpr int Md = 2, Mh = 7, Mw = 7;
constexpr int Sd = 1, Sh = 3, Sw = 3;
constexpr int NH = 6, HD = 32;
constexpr int NT = 98, NP = 112;                 // window tokens, padded to 7*16
constexpr int NWIN = 1024;                       // B * 256 windows
constexpr float SCALE = 0.17677669529663687f;    // HD^-0.5

// ---- workspace layout (bytes) ----
constexpr size_t SZ_QH   = (size_t)NWIN * NH * NP * HD * 2;      // 44,040,192
constexpr size_t OFF_Q   = 0;                                    // bf16 [win][nh][NP][HD] (pre-scaled)
constexpr size_t OFF_K   = OFF_Q + SZ_QH;                        // bf16 [win][nh][NP][HD]
constexpr size_t OFF_VT  = OFF_K + SZ_QH;                        // bf16 [win][nh][HD][NP]
constexpr size_t OFF_O   = OFF_VT + SZ_QH;                       // bf16 [win][NP][C]
constexpr size_t OFF_H1  = OFF_O + (size_t)NWIN * NP * cC * 2;   // bf16 [B*L][C]
constexpr size_t OFF_WQ  = OFF_H1 + (size_t)cB * cL * cC * 2;    // bf16 qkv_w [576][192]
constexpr size_t OFF_WP  = OFF_WQ + (size_t)3 * cC * cC * 2;     // bf16 proj_w [192][192]
constexpr size_t OFF_WF  = OFF_WP + (size_t)cC * cC * 2;         // bf16 fc1_w  [192][192]
constexpr size_t OFF_BIAS= OFF_WF + (size_t)cC * cC * 2;         // f32 [NH][98][98]

__device__ __forceinline__ bf16 f2bf(float f) {
  unsigned u = __float_as_uint(f);
  u += 0x7FFFu + ((u >> 16) & 1u);          // round-to-nearest-even
  unsigned short h = (unsigned short)(u >> 16);
  bf16 r; __builtin_memcpy(&r, &h, 2); return r;
}

__device__ __forceinline__ v16bf ld16(const bf16* p0, const bf16* p1) {
  union { v16bf v; v8bf h[2]; } u;
  u.h[0] = *(const v8bf*)p0;
  u.h[1] = *(const v8bf*)p1;
  return u.v;
}

__device__ __forceinline__ v8f zero8() {
  v8f z = {0.f,0.f,0.f,0.f,0.f,0.f,0.f,0.f}; return z;
}

__device__ __forceinline__ v8f wmma_bf(v16bf a, v16bf b, v8f c) {
  return __builtin_amdgcn_wmma_f32_16x16x32_bf16(false, a, false, b, (short)0, c, false, false);
}

// ---------------- K0: weight bf16 conversion + rel-pos bias table ----------------
__global__ void k_prep(const float* __restrict__ qkv_w, const float* __restrict__ proj_w,
                       const float* __restrict__ fc1_w, const float* __restrict__ rel_bias,
                       char* __restrict__ ws) {
  int tid = blockIdx.x * blockDim.x + threadIdx.x;
  bf16* wq = (bf16*)(ws + OFF_WQ);
  bf16* wp = (bf16*)(ws + OFF_WP);
  bf16* wf = (bf16*)(ws + OFF_WF);
  float* bias = (float*)(ws + OFF_BIAS);
  if (tid < 3 * cC * cC) wq[tid] = f2bf(qkv_w[tid]);
  if (tid < cC * cC) { wp[tid] = f2bf(proj_w[tid]); wf[tid] = f2bf(fc1_w[tid]); }
  if (tid < NH * NT * NT) {
    int nh = tid / (NT * NT);
    int r  = (tid / NT) % NT;
    int c  = tid % NT;
    int rd = r / 49, rh = (r % 49) / 7, rw = r % 7;
    int cd = c / 49, ch = (c % 49) / 7, cw = c % 7;
    int idx = (rd - cd + Md - 1) * (2*Mh-1) * (2*Mw-1)
            + (rh - ch + Mh - 1) * (2*Mw-1)
            + (rw - cw + Mw - 1);
    bias[tid] = rel_bias[idx * NH + nh];
  }
}

// ---------------- K1: LayerNorm + shift + partition + QKV GEMM ----------------
__global__ __launch_bounds__(192) void k_qkv(const float* __restrict__ x,
                                             const float* __restrict__ ng,
                                             const float* __restrict__ nb,
                                             const float* __restrict__ qkv_b,
                                             char* __restrict__ ws) {
  __shared__ bf16 sA[NP * 200];   // [112][200] bf16, 400B row stride (16B aligned)
  int wi = blockIdx.x;
  int b = wi >> 8, rem = wi & 255;
  int dq = rem >> 6, hq = (rem >> 3) & 7, wqi = rem & 7;
  int lane = threadIdx.x & 31, wave = threadIdx.x >> 5;

  // stage 1: LN + gather (roll(-S) + window partition) -> LDS bf16
  for (int t = wave; t < NT; t += 6) {
    int md = t / 49, t2 = t % 49, mh = t2 / 7, mw = t2 % 7;
    int ds = (dq*Md + md + Sd) % cD;
    int hs = (hq*Mh + mh + Sh) % cH;
    int wsr = (wqi*Mw + mw + Sw) % cW;
    const float* row = x + ((size_t)b * cL + (size_t)(ds*cH + hs)*cW + wsr) * cC;
    float v[6], s = 0.f, ss = 0.f;
    for (int j = 0; j < 6; j++) { float f = row[lane + 32*j]; v[j] = f; s += f; ss += f*f; }
    for (int m = 16; m >= 1; m >>= 1) { s += __shfl_xor(s, m, 32); ss += __shfl_xor(ss, m, 32); }
    float mu = s * (1.f/192.f);
    float inv = rsqrtf(ss*(1.f/192.f) - mu*mu + 1e-5f);
    for (int j = 0; j < 6; j++) {
      int c = lane + 32*j;
      sA[t*200 + c] = f2bf((v[j]-mu)*inv*ng[c] + nb[c]);
    }
  }
  for (int t = NT + wave; t < NP; t += 6)
    for (int j = 0; j < 6; j++) sA[t*200 + lane + 32*j] = f2bf(0.f);
  __syncthreads();

  // stage 2: [112 x 192] x [192 x 576] GEMM via WMMA bf16
  const bf16* wqb = (const bf16*)(ws + OFF_WQ);
  bf16* qb  = (bf16*)(ws + OFF_Q);
  bf16* kb  = (bf16*)(ws + OFF_K);
  bf16* vtb = (bf16*)(ws + OFF_VT);
  int col = lane & 15, gg = lane >> 4;

  for (int nt = wave*6; nt < wave*6 + 6; nt++) {
    int n = nt*16 + col;
    v8f acc[7];
    for (int i = 0; i < 7; i++) acc[i] = zero8();
    for (int kc = 0; kc < 6; kc++) {
      const bf16* bp = wqb + (size_t)n*cC + kc*32 + gg*8;       // B: k = (e/8)*16 + g*8 + e%8
      v16bf bfrag = ld16(bp, bp + 16);
#pragma unroll
      for (int mt = 0; mt < 7; mt++) {
        const bf16* ap = &sA[(mt*16 + col)*200 + kc*32 + gg*8]; // A: k = (e/8)*16 + g*8 + e%8
        v16bf afrag = ld16(ap, ap + 16);
        acc[mt] = wmma_bf(afrag, bfrag, acc[mt]);
      }
    }
    float bia = qkv_b[n];
    for (int mt = 0; mt < 7; mt++)
      for (int v = 0; v < 8; v++) {
        int m = mt*16 + v + 8*gg;                               // C/D: row = v + 8*(lane/16)
        if (m >= NT) continue;
        float val = acc[mt][v] + bia;
        if (n < cC) {
          int head = n >> 5, hd = n & 31;
          qb[((size_t)(wi*NH + head)*NP + m)*HD + hd] = f2bf(val * SCALE);
        } else if (n < 2*cC) {
          int c2 = n - cC, head = c2 >> 5, hd = c2 & 31;
          kb[((size_t)(wi*NH + head)*NP + m)*HD + hd] = f2bf(val);
        } else {
          int c2 = n - 2*cC, head = c2 >> 5, hd = c2 & 31;
          vtb[((size_t)(wi*NH + head)*HD + hd)*NP + m] = f2bf(val);
        }
      }
  }
}

// ---------------- K2: attention per (window, head) ----------------
__global__ __launch_bounds__(224) void k_attn(const float* __restrict__ attn_mask,
                                              char* __restrict__ ws) {
  __shared__ float sS[NP * 113];   // scores f32
  __shared__ bf16  sP[NP * 136];   // probabilities bf16, K padded to 128
  int blk = blockIdx.x;
  int wi = blk / NH, nh = blk % NH;
  int lane = threadIdx.x & 31, wave = threadIdx.x >> 5;
  int col = lane & 15, gg = lane >> 4;
  const bf16* qb  = (const bf16*)(ws + OFF_Q)  + (size_t)(wi*NH + nh)*NP*HD;
  const bf16* kb  = (const bf16*)(ws + OFF_K)  + (size_t)(wi*NH + nh)*NP*HD;
  const bf16* vtb = (const bf16*)(ws + OFF_VT) + (size_t)(wi*NH + nh)*HD*NP;
  const float* bias = (const float*)(ws + OFF_BIAS) + (size_t)nh*NT*NT;

  // stage A: S = q @ k^T (q pre-scaled), K = HD = 32 -> 1 WMMA per tile
  int mt = wave;  // 7 waves = 7 M-tiles
  {
    int row = mt*16 + col;
    const bf16* ap = qb + (size_t)row*HD + gg*8;
    v16bf afrag = ld16(ap, ap + 16);
#pragma unroll
    for (int nt = 0; nt < 7; nt++) {
      int tok = nt*16 + col;
      const bf16* bp = kb + (size_t)tok*HD + gg*8;
      v16bf bfrag = ld16(bp, bp + 16);
      v8f acc = wmma_bf(afrag, bfrag, zero8());
      for (int v = 0; v < 8; v++)
        sS[(mt*16 + v + 8*gg)*113 + nt*16 + col] = acc[v];
    }
  }
  __syncthreads();

  // stage B: + rel bias + shift mask, online softmax, P -> LDS bf16 (zero-padded)
  int r = threadIdx.x;
  if (r < NP) {
    bf16* prow = &sP[r * 136];
    if (r < NT) {
      int widx = wi & 255;
      const float* br = bias + (size_t)r * NT;
      const float* mr = attn_mask + ((size_t)widx * NT + r) * NT;
      float* srow = &sS[r * 113];
      float mmax = -1e30f, ssum = 0.f;
      for (int c = 0; c < NT; c++) {
        float val = srow[c] + br[c] + mr[c];
        srow[c] = val;
        if (val > mmax) { ssum = ssum * __expf(mmax - val) + 1.f; mmax = val; }
        else ssum += __expf(val - mmax);
      }
      float inv = 1.f / ssum;
      for (int c = 0; c < NT; c++) prow[c] = f2bf(__expf(srow[c] - mmax) * inv);
      for (int c = NT; c < 128; c++) prow[c] = f2bf(0.f);
    } else {
      for (int c = 0; c < 128; c++) prow[c] = f2bf(0.f);
    }
  }
  __syncthreads();

  // stage C: O = P @ v  (K padded to 128 -> 4 WMMA k-steps)
  bf16* ob = (bf16*)(ws + OFF_O) + (size_t)wi * NP * cC;
#pragma unroll
  for (int nt2 = 0; nt2 < 2; nt2++) {
    v8f acc = zero8();
#pragma unroll
    for (int kc = 0; kc < 4; kc++) {
      const bf16* ap = &sP[(mt*16 + col)*136 + kc*32 + gg*8];
      v16bf afrag = ld16(ap, ap + 16);
      int hd = nt2*16 + col;
      const bf16* bp = vtb + (size_t)hd*NP + kc*32 + gg*8;
      v16bf bfrag = ld16(bp, bp + 16);
      acc = wmma_bf(afrag, bfrag, acc);
    }
    for (int v = 0; v < 8; v++) {
      int m = mt*16 + v + 8*gg;
      if (m >= NT) continue;
      ob[(size_t)m*cC + nh*HD + nt2*16 + col] = f2bf(acc[v]);
    }
  }
}

// ---------------- K3: proj GEMM + window reverse + roll(+S) scatter ----------------
__global__ __launch_bounds__(192) void k_proj(const float* __restrict__ proj_b,
                                              char* __restrict__ ws) {
  int wi = blockIdx.x;
  int b = wi >> 8, rem = wi & 255;
  int dq = rem >> 6, hq = (rem >> 3) & 7, wqi = rem & 7;
  int lane = threadIdx.x & 31, wave = threadIdx.x >> 5;
  int col = lane & 15, gg = lane >> 4;
  const bf16* ob  = (const bf16*)(ws + OFF_O) + (size_t)wi * NP * cC;
  const bf16* wpb = (const bf16*)(ws + OFF_WP);
  bf16* h1 = (bf16*)(ws + OFF_H1);

  for (int nt = wave*2; nt < wave*2 + 2; nt++) {
    int n = nt*16 + col;
    v8f acc[7];
    for (int i = 0; i < 7; i++) acc[i] = zero8();
    for (int kc = 0; kc < 6; kc++) {
      const bf16* bp = wpb + (size_t)n*cC + kc*32 + gg*8;
      v16bf bfrag = ld16(bp, bp + 16);
#pragma unroll
      for (int mt = 0; mt < 7; mt++) {
        const bf16* ap = ob + (size_t)(mt*16 + col)*cC + kc*32 + gg*8;
        v16bf afrag = ld16(ap, ap + 16);
        acc[mt] = wmma_bf(afrag, bfrag, acc[mt]);
      }
    }
    float pb = proj_b[n];
    for (int mt = 0; mt < 7; mt++)
      for (int v = 0; v < 8; v++) {
        int m = mt*16 + v + 8*gg;
        if (m >= NT) continue;
        int md = m / 49, t2 = m % 49, mh = t2 / 7, mw = t2 % 7;
        int df = (dq*Md + md + Sd) % cD;
        int hf = (hq*Mh + mh + Sh) % cH;
        int wf = (wqi*Mw + mw + Sw) % cW;
        size_t pos = (size_t)b * cL + (size_t)(df*cH + hf)*cW + wf;
        h1[pos*cC + n] = f2bf(acc[mt][v] + pb);
      }
  }
}

// ---------------- K4: FC1 GEMM + exact GELU + residual ----------------
__global__ __launch_bounds__(192) void k_mlp(const float* __restrict__ x,
                                             const float* __restrict__ fc1_b,
                                             float* __restrict__ out,
                                             const char* __restrict__ ws) {
  int blk = blockIdx.x;                       // 784 blocks x 128 rows
  int lane = threadIdx.x & 31, wave = threadIdx.x >> 5;
  int col = lane & 15, gg = lane >> 4;
  const bf16* h1  = (const bf16*)(ws + OFF_H1);
  const bf16* wfb = (const bf16*)(ws + OFF_WF);
  size_t row0 = (size_t)blk * 128;

  for (int nt = wave*2; nt < wave*2 + 2; nt++) {
    int n = nt*16 + col;
    v8f acc[8];
    for (int i = 0; i < 8; i++) acc[i] = zero8();
    for (int kc = 0; kc < 6; kc++) {
      const bf16* bp = wfb + (size_t)n*cC + kc*32 + gg*8;
      v16bf bfrag = ld16(bp, bp + 16);
#pragma unroll
      for (int mt = 0; mt < 8; mt++) {
        const bf16* ap = h1 + (row0 + mt*16 + col)*cC + kc*32 + gg*8;
        v16bf afrag = ld16(ap, ap + 16);
        acc[mt] = wmma_bf(afrag, bfrag, acc[mt]);
      }
    }
    float fb = fc1_b[n];
    for (int mt = 0; mt < 8; mt++)
      for (int v = 0; v < 8; v++) {
        size_t m = row0 + mt*16 + v + 8*gg;
        float val = acc[mt][v] + fb;
        float ge = 0.5f * val * (1.f + erff(val * 0.70710678118654752f));
        out[m*cC + n] = x[m*cC + n] + ge;
      }
  }
}

extern "C" void kernel_launch(void* const* d_in, const int* in_sizes, int n_in,
                              void* d_out, int out_size, void* d_ws, size_t ws_size,
                              hipStream_t stream) {
  (void)in_sizes; (void)n_in; (void)out_size; (void)ws_size;
  const float* x         = (const float*)d_in[0];
  const float* attn_mask = (const float*)d_in[1];
  const float* norm1_g   = (const float*)d_in[2];
  const float* norm1_b   = (const float*)d_in[3];
  const float* qkv_w     = (const float*)d_in[4];
  const float* qkv_b     = (const float*)d_in[5];
  const float* rel_bias  = (const float*)d_in[6];
  const float* proj_w    = (const float*)d_in[7];
  const float* proj_b    = (const float*)d_in[8];
  const float* fc1_w     = (const float*)d_in[9];
  const float* fc1_b     = (const float*)d_in[10];
  char* ws = (char*)d_ws;
  float* out = (float*)d_out;

  hipLaunchKernelGGL(k_prep, dim3((3*cC*cC + 255)/256), dim3(256), 0, stream,
                     qkv_w, proj_w, fc1_w, rel_bias, ws);
  hipLaunchKernelGGL(k_qkv, dim3(NWIN), dim3(192), 0, stream,
                     x, norm1_g, norm1_b, qkv_b, ws);
  hipLaunchKernelGGL(k_attn, dim3(NWIN*NH), dim3(224), 0, stream, attn_mask, ws);
  hipLaunchKernelGGL(k_proj, dim3(NWIN), dim3(192), 0, stream, proj_b, ws);
  hipLaunchKernelGGL(k_mlp, dim3((cB*cL)/128), dim3(192), 0, stream,
                     x, fc1_b, out, ws);
}
